// LocalGlobalTemporalEncoder_33182917328923
// MI455X (gfx1250) — compile-verified, matrix operand order checked by reading, not run on hardware
//
#include <hip/hip_runtime.h>

// ---------------------------------------------------------------------------
// LocalGlobalTemporalEncoder for MI455X (gfx1250, wave32, WMMA)
// B=8, C=512, T=1024, H=8, HD=64, DFF=256, WIN=9 (band +-4 on first 4 heads)
// Heavy math: v_wmma_f32_16x16x32_bf16 (fp32 accumulate).
// GEMMs stage tiles via async global->LDS copies (ASYNCcnt) with double buffer;
// each wave computes a 32x64 strip (8 WMMAs per k-step, B-frags reused).
// ---------------------------------------------------------------------------

#define BB 8
#define CC 512
#define TT 1024
#define HH 8
#define HD 64
#define DFF 256
#define QBLK 64            // T/16 query blocks per (b,h)
#define EPS 1e-5f

typedef __attribute__((ext_vector_type(16))) __bf16 bf16x16;
typedef __attribute__((ext_vector_type(8)))  __bf16 bf16x8;
typedef __attribute__((ext_vector_type(8)))  float  f32x8;

// pointer types for the async global->LDS builtin: (v4i AS1*, v4i AS3*, Ii, Ii)
typedef int v4i __attribute__((vector_size(16)));
typedef v4i __attribute__((address_space(1)))* gv4i_p;
typedef v4i __attribute__((address_space(3)))* lv4i_p;

__device__ __forceinline__ bf16x16 ldfrag(const __bf16* p) {
    // 16 bf16 as two 16B chunks at p and p+16 elements.
    bf16x8 lo = *(const bf16x8*)(p);
    bf16x8 hi = *(const bf16x8*)(p + 16);
    return __builtin_shufflevector(lo, hi, 0,1,2,3,4,5,6,7,8,9,10,11,12,13,14,15);
}

__device__ __forceinline__ f32x8 wmma_bf16(bf16x16 a, bf16x16 b, f32x8 c) {
    return __builtin_amdgcn_wmma_f32_16x16x32_bf16(
        /*neg_a=*/false, a, /*neg_b=*/false, b,
        /*c_mod=*/(short)0, c, /*reuse_a=*/false, /*reuse_b=*/false);
}

// ---- async global->LDS copy (16B per op), with safe fallback -------------
__device__ __forceinline__ void cp_async16(__bf16* l, const __bf16* g) {
#if defined(__has_builtin) && __has_builtin(__builtin_amdgcn_global_load_async_to_lds_b128)
    __builtin_amdgcn_global_load_async_to_lds_b128((gv4i_p)g, (lv4i_p)l, 0, 0);
#else
    *(bf16x8*)l = *(const bf16x8*)g;                // sync fallback
#endif
}
__device__ __forceinline__ void cp_wait6() {
#if defined(__has_builtin) && __has_builtin(__builtin_amdgcn_s_wait_asynccnt)
    __builtin_amdgcn_s_wait_asynccnt(6);
#endif
}
__device__ __forceinline__ void cp_wait0() {
#if defined(__has_builtin) && __has_builtin(__builtin_amdgcn_s_wait_asynccnt)
    __builtin_amdgcn_s_wait_asynccnt(0);
#endif
}

// ---------------------------------------------------------------------------
// Pre-pass conversions
// ---------------------------------------------------------------------------
__global__ __launch_bounds__(256) void k_feat_to_bf16(const float* __restrict__ f,
                                                      __bf16* __restrict__ srcT) {
    // f: [B,C,T] fp32  ->  srcT: [B,T,C] bf16
    int idx = blockIdx.x * 256 + threadIdx.x;          // over B*T*C
    int c = idx & (CC - 1);
    int t = (idx >> 9) & (TT - 1);
    int b = idx >> 19;
    srcT[idx] = (__bf16)f[((size_t)b * CC + c) * TT + t];
}

__global__ __launch_bounds__(256) void k_f32_to_bf16(const float* __restrict__ in,
                                                     __bf16* __restrict__ out, int n) {
    int idx = blockIdx.x * 256 + threadIdx.x;
    if (idx < n) out[idx] = (__bf16)in[idx];
}

__global__ __launch_bounds__(256) void k_extract_vT(const __bf16* __restrict__ qkv,
                                                    __bf16* __restrict__ vT) {
    // qkv: [B,T,3C] bf16 ; vT: [B,H,HD,T] bf16 (contiguous in t)
    int idx = blockIdx.x * 256 + threadIdx.x;          // over B*H*HD*T
    int t = idx & (TT - 1);
    int r = idx >> 10;
    int d = r & (HD - 1);
    int h = (r >> 6) & (HH - 1);
    int b = r >> 9;
    vT[idx] = qkv[((size_t)(b * TT + t)) * (3 * CC) + 2 * CC + h * HD + d];
}

// ---------------------------------------------------------------------------
// WMMA GEMM with async-LDS double buffering:
//   out[M,N] = A[M,K](bf16) * W[N,K]^T(bf16) + bias, optional ReLU.
// Block = 128 threads (4 waves) owns a 128x64 output tile; each wave a 32x64
// strip.  Per k-step: stage A[128x32] + B[64x32] to LDS asynchronously (12KB),
// 8 WMMAs per wave (B fragments reused across both row groups).
// M%128==0, N%64==0, K%32==0.
// ---------------------------------------------------------------------------
template <int OUT_BF16, int RELU>
__global__ __launch_bounds__(128) void k_gemm(const __bf16* __restrict__ A,
                                              const __bf16* __restrict__ W,
                                              const float* __restrict__ bias,
                                              void* __restrict__ out,
                                              int M, int N, int K) {
    const int tid  = threadIdx.x;
    const int lane = tid & 31;
    const int wave = tid >> 5;
    const int lm   = lane & 15;
    const int half = lane >> 4;
    const int row0 = blockIdx.x * 128;
    const int n0   = blockIdx.y * 64;

    __shared__ __attribute__((aligned(16))) __bf16 sA[2][128 * 32];
    __shared__ __attribute__((aligned(16))) __bf16 sB[2][64 * 32];

    const int NK = K >> 5;

    // stage one 128x32 A tile + 64x32 B tile: 6 x 16B async ops per thread
    auto stage = [&](int buf, int kk) {
        const int k0 = kk * 32;
#pragma unroll
        for (int i = 0; i < 4; ++i) {
            int c = tid + i * 128;             // 0..511 chunks of 16B
            int r = c >> 2;
            int s = (c & 3) * 8;
            cp_async16(&sA[buf][r * 32 + s], A + (size_t)(row0 + r) * K + k0 + s);
        }
#pragma unroll
        for (int i = 0; i < 2; ++i) {
            int c = tid + i * 128;             // 0..255 chunks of 16B
            int r = c >> 2;
            int s = (c & 3) * 8;
            cp_async16(&sB[buf][r * 32 + s], W + (size_t)(n0 + r) * K + k0 + s);
        }
    };

    f32x8 acc0[4] = {};
    f32x8 acc1[4] = {};
    stage(0, 0);
    for (int kk = 0; kk < NK; ++kk) {
        const int buf = kk & 1;
        if (kk + 1 < NK) { stage(buf ^ 1, kk + 1); cp_wait6(); }
        else             { cp_wait0(); }
        __syncthreads();                       // tile kk visible to all waves
        bf16x16 af0 = ldfrag(&sA[buf][(wave * 32 + lm) * 32 + half * 8]);
        bf16x16 af1 = ldfrag(&sA[buf][(wave * 32 + 16 + lm) * 32 + half * 8]);
#pragma unroll
        for (int nt = 0; nt < 4; ++nt) {
            bf16x16 bf_ = ldfrag(&sB[buf][(nt * 16 + lm) * 32 + half * 8]);
            acc0[nt] = wmma_bf16(af0, bf_, acc0[nt]);
            acc1[nt] = wmma_bf16(af1, bf_, acc1[nt]);
        }
        __syncthreads();                       // all reads done before reuse
    }

#pragma unroll
    for (int nt = 0; nt < 4; ++nt) {
        float bv = bias[n0 + nt * 16 + lm];
        const int col = n0 + nt * 16 + lm;
#pragma unroll
        for (int r = 0; r < 8; ++r) {
            float v0 = acc0[nt][r] + bv;
            float v1 = acc1[nt][r] + bv;
            if (RELU) { v0 = fmaxf(v0, 0.0f); v1 = fmaxf(v1, 0.0f); }
            size_t o0 = (size_t)(row0 + wave * 32 + r + 8 * half) * N + col;
            size_t o1 = (size_t)(row0 + wave * 32 + 16 + r + 8 * half) * N + col;
            if (OUT_BF16) {
                ((__bf16*)out)[o0] = (__bf16)v0;
                ((__bf16*)out)[o1] = (__bf16)v1;
            } else {
                ((float*)out)[o0] = v0;
                ((float*)out)[o1] = v1;
            }
        }
    }
}

// ---------------------------------------------------------------------------
// Flash-style attention. One wave per (b, h, 16-query block).
// Local heads (h < 4): band |q-k| <= 4 -> only chunks overlapping the band.
// ---------------------------------------------------------------------------
__global__ __launch_bounds__(128) void k_attn(const __bf16* __restrict__ qkv,
                                              const __bf16* __restrict__ vT,
                                              __bf16* __restrict__ ctx) {
    const int lane = threadIdx.x & 31;
    const int wave = threadIdx.x >> 5;
    const int lm   = lane & 15;
    const int half = lane >> 4;

    const int wid = blockIdx.x * 4 + wave;     // 0 .. B*H*QBLK-1
    const int qb  = wid & (QBLK - 1);
    const int h   = (wid >> 6) & (HH - 1);
    const int b   = wid >> 9;
    const int q0  = qb * 16;
    const bool local = (h < HH / 2);

    __shared__ __attribute__((aligned(16))) __bf16 plds[4][16 * 32];
    __bf16* my = plds[wave];

    // Q fragments (A-matrix, 16x64 -> two 16x32 k-steps)
    const __bf16* qp = qkv + ((size_t)(b * TT + q0 + lm)) * (3 * CC) + h * HD + half * 8;
    bf16x16 aq0 = ldfrag(qp);
    bf16x16 aq1 = ldfrag(qp + 32);

    f32x8 o0 = {}, o1 = {}, o2 = {}, o3 = {};
    float m_run[8], l_run[8];
#pragma unroll
    for (int r = 0; r < 8; ++r) { m_run[r] = -3.0e38f; l_run[r] = 0.0f; }

    int c_lo = 0, c_hi = (TT / 32) - 1;
    if (local) {
        int lo = q0 - 4; if (lo < 0) lo = 0;
        int hi = q0 + 19; if (hi > TT - 1) hi = TT - 1;
        c_lo = lo >> 5; c_hi = hi >> 5;
    }

    for (int c = c_lo; c <= c_hi; ++c) {
        const int kc = c * 32;
        // ---- scores: S[16 x 32] = Q * K^T, two 16-key column tiles
        f32x8 s0 = {}, s1 = {};
        {
            const __bf16* kp = qkv + ((size_t)(b * TT + kc + lm)) * (3 * CC)
                                   + CC + h * HD + half * 8;
            s0 = wmma_bf16(aq0, ldfrag(kp), s0);
            s0 = wmma_bf16(aq1, ldfrag(kp + 32), s0);
            const __bf16* kp1 = kp + (size_t)16 * 3 * CC;
            s1 = wmma_bf16(aq0, ldfrag(kp1), s1);
            s1 = wmma_bf16(aq1, ldfrag(kp1 + 32), s1);
        }
        // ---- scale + band mask
#pragma unroll
        for (int r = 0; r < 8; ++r) {
            int qi = q0 + r + 8 * half;
            float v0 = s0[r] * 0.125f;          // HD^-0.5
            float v1 = s1[r] * 0.125f;
            if (local) {
                int d0 = qi - (kc + lm);        if (d0 < 0) d0 = -d0;
                int d1 = qi - (kc + 16 + lm);   if (d1 < 0) d1 = -d1;
                if (d0 > 4) v0 = -1.0e30f;
                if (d1 > 4) v1 = -1.0e30f;
            }
            s0[r] = v0; s1[r] = v1;
        }
        // ---- online softmax (rows live across 16-lane groups)
        float cmax[8];
#pragma unroll
        for (int r = 0; r < 8; ++r) cmax[r] = fmaxf(s0[r], s1[r]);
#pragma unroll
        for (int off = 8; off; off >>= 1)
#pragma unroll
            for (int r = 0; r < 8; ++r)
                cmax[r] = fmaxf(cmax[r], __shfl_xor(cmax[r], off, 32));

        float esc[8], csum[8];
#pragma unroll
        for (int r = 0; r < 8; ++r) {
            float mn = fmaxf(m_run[r], cmax[r]);
            esc[r] = __expf(m_run[r] - mn);
            m_run[r] = mn;
            float p0 = __expf(s0[r] - mn);
            float p1 = __expf(s1[r] - mn);
            s0[r] = p0; s1[r] = p1;
            csum[r] = p0 + p1;
        }
#pragma unroll
        for (int off = 8; off; off >>= 1)
#pragma unroll
            for (int r = 0; r < 8; ++r)
                csum[r] += __shfl_xor(csum[r], off, 32);
#pragma unroll
        for (int r = 0; r < 8; ++r) {
            l_run[r] = l_run[r] * esc[r] + csum[r];
            o0[r] *= esc[r]; o1[r] *= esc[r]; o2[r] *= esc[r]; o3[r] *= esc[r];
        }
        // ---- transpose P through wave-private LDS (DS ops in-order in wave)
#pragma unroll
        for (int r = 0; r < 8; ++r) {
            int m = r + 8 * half;
            my[m * 32 + lm]      = (__bf16)s0[r];
            my[m * 32 + 16 + lm] = (__bf16)s1[r];
        }
        bf16x16 ap = ldfrag(my + lm * 32 + half * 8);
        // ---- O += P * V   (V from transposed vT, contiguous in t)
        const __bf16* vp = vT + ((size_t)(b * HH + h) * HD + lm) * TT + kc + half * 8;
        o0 = wmma_bf16(ap, ldfrag(vp), o0);
        o1 = wmma_bf16(ap, ldfrag(vp + (size_t)16 * TT), o1);
        o2 = wmma_bf16(ap, ldfrag(vp + (size_t)32 * TT), o2);
        o3 = wmma_bf16(ap, ldfrag(vp + (size_t)48 * TT), o3);
    }

    // ---- normalize + store ctx[B,T,C] bf16
#pragma unroll
    for (int r = 0; r < 8; ++r) {
        float inv = 1.0f / l_run[r];
        int qi = q0 + r + 8 * half;
        size_t base = ((size_t)(b * TT + qi)) * CC + h * HD;
        ctx[base + lm]      = (__bf16)(o0[r] * inv);
        ctx[base + 16 + lm] = (__bf16)(o1[r] * inv);
        ctx[base + 32 + lm] = (__bf16)(o2[r] * inv);
        ctx[base + 48 + lm] = (__bf16)(o3[r] * inv);
    }
}

// ---------------------------------------------------------------------------
// Residual + LayerNorm. LN1: residual = transposed features, emits f32+bf16.
// LN2: residual = ln1 f32, emits final transposed output [B,C,T].
// ---------------------------------------------------------------------------
__global__ __launch_bounds__(256) void k_add_ln1(const float* __restrict__ feats,
                                                 const float* __restrict__ y,
                                                 const float* __restrict__ g,
                                                 const float* __restrict__ be,
                                                 float* __restrict__ ln_f32,
                                                 __bf16* __restrict__ ln_bf16) {
    const int bt = blockIdx.x;                 // B*T rows
    const int b = bt >> 10, t = bt & (TT - 1);
    const int tid = threadIdx.x;
    __shared__ float red[256];

    float x0 = feats[((size_t)b * CC + tid) * TT + t]       + y[(size_t)bt * CC + tid];
    float x1 = feats[((size_t)b * CC + tid + 256) * TT + t] + y[(size_t)bt * CC + tid + 256];

    red[tid] = x0 + x1; __syncthreads();
    for (int off = 128; off; off >>= 1) { if (tid < off) red[tid] += red[tid + off]; __syncthreads(); }
    float mu = red[0] * (1.0f / CC);
    __syncthreads();
    float d0 = x0 - mu, d1 = x1 - mu;
    red[tid] = d0 * d0 + d1 * d1; __syncthreads();
    for (int off = 128; off; off >>= 1) { if (tid < off) red[tid] += red[tid + off]; __syncthreads(); }
    float rs = rsqrtf(red[0] * (1.0f / CC) + EPS);

    float o0 = d0 * rs * g[tid]       + be[tid];
    float o1 = d1 * rs * g[tid + 256] + be[tid + 256];
    ln_f32[(size_t)bt * CC + tid]        = o0;
    ln_f32[(size_t)bt * CC + tid + 256]  = o1;
    ln_bf16[(size_t)bt * CC + tid]       = (__bf16)o0;
    ln_bf16[(size_t)bt * CC + tid + 256] = (__bf16)o1;
}

__global__ __launch_bounds__(256) void k_add_ln2(const float* __restrict__ ln1,
                                                 const float* __restrict__ ff,
                                                 const float* __restrict__ g,
                                                 const float* __restrict__ be,
                                                 float* __restrict__ out) {
    const int bt = blockIdx.x;
    const int b = bt >> 10, t = bt & (TT - 1);
    const int tid = threadIdx.x;
    __shared__ float red[256];

    float x0 = ln1[(size_t)bt * CC + tid]       + ff[(size_t)bt * CC + tid];
    float x1 = ln1[(size_t)bt * CC + tid + 256] + ff[(size_t)bt * CC + tid + 256];

    red[tid] = x0 + x1; __syncthreads();
    for (int off = 128; off; off >>= 1) { if (tid < off) red[tid] += red[tid + off]; __syncthreads(); }
    float mu = red[0] * (1.0f / CC);
    __syncthreads();
    float d0 = x0 - mu, d1 = x1 - mu;
    red[tid] = d0 * d0 + d1 * d1; __syncthreads();
    for (int off = 128; off; off >>= 1) { if (tid < off) red[tid] += red[tid + off]; __syncthreads(); }
    float rs = rsqrtf(red[0] * (1.0f / CC) + EPS);

    out[((size_t)b * CC + tid) * TT + t]       = d0 * rs * g[tid]       + be[tid];
    out[((size_t)b * CC + tid + 256) * TT + t] = d1 * rs * g[tid + 256] + be[tid + 256];
}

// ---------------------------------------------------------------------------
// Host launcher
// ---------------------------------------------------------------------------
extern "C" void kernel_launch(void* const* d_in, const int* in_sizes, int n_in,
                              void* d_out, int out_size, void* d_ws, size_t ws_size,
                              hipStream_t stream) {
    const float* features  = (const float*)d_in[0];   // [B,C,T]
    const float* in_proj_w = (const float*)d_in[1];   // [3C,C]
    const float* in_proj_b = (const float*)d_in[2];   // [3C]
    const float* out_w     = (const float*)d_in[3];   // [C,C]
    const float* out_b     = (const float*)d_in[4];   // [C]
    const float* w1        = (const float*)d_in[5];   // [DFF,C]
    const float* b1        = (const float*)d_in[6];   // [DFF]
    const float* w2        = (const float*)d_in[7];   // [C,DFF]
    const float* b2        = (const float*)d_in[8];   // [C]
    const float* g1        = (const float*)d_in[9];
    const float* beta1     = (const float*)d_in[10];
    const float* g2        = (const float*)d_in[11];
    const float* beta2     = (const float*)d_in[12];
    float* out = (float*)d_out;

    const size_t NTOK  = (size_t)BB * TT;        // 8192
    const size_t S_BTC = NTOK * CC;              // 4,194,304 elements

    // workspace carve-up (bytes)
    char* ws = (char*)d_ws;
    size_t off = 0;
    auto alloc = [&](size_t bytes) { char* p = ws + off; off += (bytes + 255) & ~(size_t)255; return p; };
    __bf16* srcT   = (__bf16*)alloc(S_BTC * 2);              // [B,T,C]
    __bf16* qkv    = (__bf16*)alloc(NTOK * 3 * CC * 2);      // [B,T,3C]
    __bf16* vT     = (__bf16*)alloc(S_BTC * 2);              // [B,H,HD,T]
    __bf16* ctx    = (__bf16*)alloc(S_BTC * 2);              // [B,T,C]
    float*  attnp  = (float*)alloc(S_BTC * 4);               // out-proj f32
    float*  ln1f   = (float*)alloc(S_BTC * 4);
    __bf16* ln1b   = (__bf16*)alloc(S_BTC * 2);
    __bf16* h1     = (__bf16*)alloc(NTOK * DFF * 2);
    float*  ffout  = (float*)alloc(S_BTC * 4);
    __bf16* wqkv   = (__bf16*)alloc((size_t)3 * CC * CC * 2);
    __bf16* wout   = (__bf16*)alloc((size_t)CC * CC * 2);
    __bf16* w1b    = (__bf16*)alloc((size_t)DFF * CC * 2);
    __bf16* w2b    = (__bf16*)alloc((size_t)CC * DFF * 2);
    (void)ws_size;

    // 1. convert inputs to bf16
    k_feat_to_bf16<<<(int)(S_BTC / 256), 256, 0, stream>>>(features, srcT);
    k_f32_to_bf16<<<(3 * CC * CC + 255) / 256, 256, 0, stream>>>(in_proj_w, wqkv, 3 * CC * CC);
    k_f32_to_bf16<<<(CC * CC + 255) / 256, 256, 0, stream>>>(out_w, wout, CC * CC);
    k_f32_to_bf16<<<(DFF * CC + 255) / 256, 256, 0, stream>>>(w1, w1b, DFF * CC);
    k_f32_to_bf16<<<(CC * DFF + 255) / 256, 256, 0, stream>>>(w2, w2b, CC * DFF);

    // 2. QKV projection: [8192,1536] = srcT[8192,512] x Wqkv^T
    k_gemm<1, 0><<<dim3((int)(NTOK / 128), (3 * CC) / 64), 128, 0, stream>>>(
        srcT, wqkv, in_proj_b, qkv, (int)NTOK, 3 * CC, CC);

    // 3. stage V transposed for attention B-fragments
    k_extract_vT<<<(int)(S_BTC / 256), 256, 0, stream>>>(qkv, vT);

    // 4. attention (banded + dense heads), one wave per 16-query block
    k_attn<<<(BB * HH * QBLK) / 4, 128, 0, stream>>>(qkv, vT, ctx);

    // 5. output projection: [8192,512] f32
    k_gemm<0, 0><<<dim3((int)(NTOK / 128), CC / 64), 128, 0, stream>>>(
        ctx, wout, out_b, attnp, (int)NTOK, CC, CC);

    // 6. residual + LN1
    k_add_ln1<<<(int)NTOK, 256, 0, stream>>>(features, attnp, g1, beta1, ln1f, ln1b);

    // 7. FFN up (ReLU): [8192,256] bf16
    k_gemm<1, 1><<<dim3((int)(NTOK / 128), DFF / 64), 128, 0, stream>>>(
        ln1b, w1b, b1, h1, (int)NTOK, DFF, CC);

    // 8. FFN down: [8192,512] f32
    k_gemm<0, 0><<<dim3((int)(NTOK / 128), CC / 64), 128, 0, stream>>>(
        h1, w2b, b2, ffout, (int)NTOK, CC, DFF);

    // 9. residual + LN2 + transpose to [B,C,T]
    k_add_ln2<<<(int)NTOK, 256, 0, stream>>>(ln1f, ffout, g2, beta2, out);
}
